// NormalizedCausalSelfAttention_72885595013416
// MI455X (gfx1250) — compile-verified, matrix operand order checked by reading, not run on hardware
//
#include <hip/hip_runtime.h>
#include <stdint.h>

// Problem constants (from reference)
#define T_SEQ 2048
#define C_DIM 2048
#define NH    16
#define HD    128

typedef __attribute__((ext_vector_type(16))) __bf16 v16bf;
typedef __attribute__((ext_vector_type(8)))  float  v8f;

union FragBF { uint4 u[2]; v16bf v; };

__device__ __forceinline__ unsigned short f2bf(float f) {
    union { float f; uint32_t u; } x; x.f = f;
    uint32_t r = x.u + 0x7FFFu + ((x.u >> 16) & 1u);  // RNE
    return (unsigned short)(r >> 16);
}

// Load a 16-element bf16 fragment (32 bytes) as two 16B chunks (works for LDS & global).
__device__ __forceinline__ v16bf ld_frag(const unsigned short* p) {
    FragBF f;
    const uint4* q = (const uint4*)p;
    f.u[0] = q[0]; f.u[1] = q[1];
    return f.v;
}

// CDNA5 async copy global->LDS (bypasses VGPRs, tracked by ASYNCcnt).
// vdst VGPR holds the LDS byte address (low 32 bits of the generic pointer),
// vaddr is the 64-bit global address.
__device__ __forceinline__ void async_ld_b128(void* lds, const void* gptr) {
#if defined(__gfx1250__)
    unsigned l = (unsigned)(uintptr_t)lds;
    unsigned long long g = (unsigned long long)(uintptr_t)gptr;
    asm volatile("global_load_async_to_lds_b128 %0, %1, off"
                 :: "v"(l), "v"(g) : "memory");
#else
    (void)lds; (void)gptr;
#endif
}
__device__ __forceinline__ void wait_async_le4() {
#if defined(__gfx1250__)
    asm volatile("s_wait_asynccnt 0x4" ::: "memory");
#endif
}

// ---------------------------------------------------------------------------
// fp32 -> bf16 bulk convert (vectorized b128 loads, b64 stores)
// ---------------------------------------------------------------------------
__global__ void cvt_f32_bf16(const float* __restrict__ src,
                             unsigned short* __restrict__ dst, int n4) {
    int i = blockIdx.x * blockDim.x + threadIdx.x;
    if (i >= n4) return;
    float4 f = ((const float4*)src)[i];
    uint2 o;
    o.x = (uint32_t)f2bf(f.x) | ((uint32_t)f2bf(f.y) << 16);
    o.y = (uint32_t)f2bf(f.z) | ((uint32_t)f2bf(f.w) << 16);
    ((uint2*)dst)[i] = o;
}

// ---------------------------------------------------------------------------
// NT GEMM: Y[m][n] = sum_k A[m][k] * W[n][k], A/W bf16, f32 accum via WMMA.
// Block tile 128x128, K-step 32, double-buffered LDS filled with
// global_load_async_to_lds_b128 (4 async instr/wave/step; s_wait_asynccnt 4
// + barrier ensures the current buffer is complete while the next is in
// flight). 8 waves: wave w = (wm = w>>1)*32 rows x (wn = w&1)*64 cols,
// 8 wmma per K-step against 12 fragment ds_load_b128s.
// Epilogue layouts:
//   layA: 1 = f32 [m][n] ("TC"), 2 = f32 [h][m][d] ("HTD", h=n>>7,d=n&127)
//   layB: 3 = bf16 [h][d][m] ("HDT")
// ---------------------------------------------------------------------------
__global__ __launch_bounds__(256)
void gemm_bf16_nt(const unsigned short* __restrict__ A,
                  const unsigned short* __restrict__ W,
                  float* __restrict__ dA, int layA,
                  unsigned short* __restrict__ dB, int layB) {
    __shared__ __align__(16) unsigned short As[2][128][40];  // 2 x 10 KB
    __shared__ __align__(16) unsigned short Bs[2][128][40];
    const int tid = threadIdx.x;
    const int w = tid >> 5, ln = tid & 31;
    const int mBase = blockIdx.x * 128;
    const int nBase = blockIdx.y * 128;
    const int wm = (w >> 1) * 32;
    const int wn = (w & 1) * 64;

    const v8f vz = {0,0,0,0,0,0,0,0};
    v8f acc[2][4];
    #pragma unroll
    for (int im = 0; im < 2; ++im)
        #pragma unroll
        for (int jn = 0; jn < 4; ++jn) acc[im][jn] = vz;

    const int fr = ln & 15;
    const int kh = (ln >> 4) * 16;

    // fill rows: each thread covers 2 uint4 per array per step
    const int r0 = tid >> 2;                 // 0..63
    const int r1 = r0 + 64;                  // 64..127
    const int kc = (tid & 3) * 8;            // 0,8,16,24

    // prologue fill of buffer 0 with k0 = 0
    async_ld_b128(&As[0][r0][kc], A + (size_t)(mBase + r0) * C_DIM + kc);
    async_ld_b128(&As[0][r1][kc], A + (size_t)(mBase + r1) * C_DIM + kc);
    async_ld_b128(&Bs[0][r0][kc], W + (size_t)(nBase + r0) * C_DIM + kc);
    async_ld_b128(&Bs[0][r1][kc], W + (size_t)(nBase + r1) * C_DIM + kc);

    for (int k0 = 0, it = 0; k0 < C_DIM; k0 += 32, ++it) {
        const int cur = it & 1;
        const int nxt = cur ^ 1;
        const int kn = (k0 + 32) & (C_DIM - 1);  // wrap on last iter (harmless refill)
        // kick off next buffer while current is consumed
        async_ld_b128(&As[nxt][r0][kc], A + (size_t)(mBase + r0) * C_DIM + kn + kc);
        async_ld_b128(&As[nxt][r1][kc], A + (size_t)(mBase + r1) * C_DIM + kn + kc);
        async_ld_b128(&Bs[nxt][r0][kc], W + (size_t)(nBase + r0) * C_DIM + kn + kc);
        async_ld_b128(&Bs[nxt][r1][kc], W + (size_t)(nBase + r1) * C_DIM + kn + kc);
        // async loads complete in order: <=4 outstanding => current buffer done
        wait_async_le4();
        __syncthreads();

        v16bf aA[2], bB[4];
        #pragma unroll
        for (int im = 0; im < 2; ++im)
            aA[im] = ld_frag(&As[cur][wm + im * 16 + fr][kh]);
        #pragma unroll
        for (int jn = 0; jn < 4; ++jn)
            bB[jn] = ld_frag(&Bs[cur][wn + jn * 16 + fr][kh]);
        #pragma unroll
        for (int im = 0; im < 2; ++im)
            #pragma unroll
            for (int jn = 0; jn < 4; ++jn)
                acc[im][jn] = __builtin_amdgcn_wmma_f32_16x16x32_bf16(
                    false, aA[im], false, bB[jn], (short)0, acc[im][jn], false, false);
        __syncthreads();  // all waves done reading cur before it is refilled
    }

    const int mhi = (ln >> 4) * 8;  // C layout: VGPR g -> M = g (+8 upper half)
    #pragma unroll
    for (int im = 0; im < 2; ++im) {
        #pragma unroll
        for (int jn = 0; jn < 4; ++jn) {
            #pragma unroll
            for (int g = 0; g < 8; ++g) {
                int m = mBase + wm + im * 16 + g + mhi;
                int n = nBase + wn + jn * 16 + fr;
                float v = acc[im][jn][g];
                if (layA == 1) {
                    dA[(size_t)m * C_DIM + n] = v;
                } else if (layA == 2) {
                    dA[((size_t)(n >> 7) * T_SEQ + m) * HD + (n & 127)] = v;
                }
                if (layB == 3) {
                    dB[((size_t)(n >> 7) * HD + (n & 127)) * T_SEQ + m] = f2bf(v);
                }
            }
        }
    }
}

// ---------------------------------------------------------------------------
// RoPE + sqk*sqrt(C) scale; reads f32 [h][t][d] pre-projections, writes bf16.
// One thread per (h, t, d<64) rotation pair.
// ---------------------------------------------------------------------------
__global__ void rope_scale(const float* __restrict__ qpre,
                           const float* __restrict__ kpre,
                           const float* __restrict__ sqk,
                           unsigned short* __restrict__ qb,
                           unsigned short* __restrict__ kbf) {
    int idx = blockIdx.x * blockDim.x + threadIdx.x;  // NH*T*64 = 2^21
    int d = idx & 63;
    int t = (idx >> 6) & (T_SEQ - 1);
    int h = idx >> 17;
    const float RESTORE = 45.254833995939045f;           // sqrt(2048)
    float inv = __expf(-(float)(2 * d) * (9.210340371976184f / 128.0f));
    float s, c;
    __sincosf((float)t * inv, &s, &c);
    size_t base = ((size_t)h * T_SEQ + t) * HD;
    float sr = sqk[h * HD + d] * RESTORE;
    float si = sqk[h * HD + d + 64] * RESTORE;
    float qr = qpre[base + d], qi = qpre[base + d + 64];
    qb[base + d]      = f2bf((qr * c - qi * s) * sr);
    qb[base + d + 64] = f2bf((qr * s + qi * c) * si);
    float kr = kpre[base + d], ki = kpre[base + d + 64];
    kbf[base + d]      = f2bf((kr * c - ki * s) * sr);
    kbf[base + d + 64] = f2bf((kr * s + ki * c) * si);
}

// ---------------------------------------------------------------------------
// Flash attention. Block = (h, 128 query rows); each wave owns 16 query rows
// and runs an independent online-softmax loop over 32-key steps (no block
// barriers — LDS P-scratch is per-wave, DS ops are in-order per wave).
//   S = Q K^T (8 wmma over D=128, two 16-key groups)   -> scale sqrt(D), mask
//   online max/sum via __shfl_xor over the 16-lane N-groups
//   P (16x32) -> bf16 -> per-wave LDS -> one A fragment
//   O += P V  (8 wmma over the 128 output dims, V stored [h][d][t])
// ---------------------------------------------------------------------------
__global__ __launch_bounds__(256)
void flash_attn(const unsigned short* __restrict__ qb,
                const unsigned short* __restrict__ kbf,
                const unsigned short* __restrict__ vt,   // [h][d][t]
                unsigned short* __restrict__ ob) {       // bf16 [t][C]
    __shared__ __align__(16) unsigned short Ps[8][16][40];
    const int w  = threadIdx.x >> 5, ln = threadIdx.x & 31;
    const int h  = blockIdx.y;
    const int qw = blockIdx.x * 128 + w * 16;
    const int fr = ln & 15;
    const int kh = (ln >> 4) * 16;
    const int mhi = (ln >> 4) * 8;

    const unsigned short* Q = qb  + (size_t)h * T_SEQ * HD;
    const unsigned short* K = kbf + (size_t)h * T_SEQ * HD;
    const unsigned short* V = vt  + (size_t)h * HD * T_SEQ;

    v16bf aQ[4];
    #pragma unroll
    for (int d4 = 0; d4 < 4; ++d4)
        aQ[d4] = ld_frag(Q + (size_t)(qw + fr) * HD + d4 * 32 + kh);

    const v8f vz = {0,0,0,0,0,0,0,0};
    v8f o[8];
    float mrow[8], lrow[8];
    #pragma unroll
    for (int i = 0; i < 8; ++i) { o[i] = vz; mrow[i] = -3.0e38f; lrow[i] = 0.f; }

    const float SC = 11.313708498984761f;  // sqrt(128)

    for (int kb0 = 0; kb0 < qw + 16; kb0 += 32) {
        v8f s0 = vz, s1 = vz;
        #pragma unroll
        for (int d4 = 0; d4 < 4; ++d4) {
            v16bf b0 = ld_frag(K + (size_t)(kb0 + fr) * HD + d4 * 32 + kh);
            v16bf b1 = ld_frag(K + (size_t)(kb0 + 16 + fr) * HD + d4 * 32 + kh);
            s0 = __builtin_amdgcn_wmma_f32_16x16x32_bf16(false, aQ[d4], false, b0,
                                                         (short)0, s0, false, false);
            s1 = __builtin_amdgcn_wmma_f32_16x16x32_bf16(false, aQ[d4], false, b1,
                                                         (short)0, s1, false, false);
        }
        float corr[8];
        #pragma unroll
        for (int g = 0; g < 8; ++g) {
            int tq = qw + g + mhi;
            float e0 = s0[g] * SC, e1 = s1[g] * SC;
            if (kb0 + fr      > tq) e0 = -3.0e38f;  // causal mask
            if (kb0 + 16 + fr > tq) e1 = -3.0e38f;
            float rm = fmaxf(e0, e1);
            rm = fmaxf(rm, __shfl_xor(rm, 1));
            rm = fmaxf(rm, __shfl_xor(rm, 2));
            rm = fmaxf(rm, __shfl_xor(rm, 4));
            rm = fmaxf(rm, __shfl_xor(rm, 8));
            float mn = fmaxf(mrow[g], rm);
            float cr = __expf(mrow[g] - mn);
            mrow[g] = mn;
            float p0 = __expf(e0 - mn);
            float p1 = __expf(e1 - mn);
            float rs = p0 + p1;
            rs += __shfl_xor(rs, 1);
            rs += __shfl_xor(rs, 2);
            rs += __shfl_xor(rs, 4);
            rs += __shfl_xor(rs, 8);
            lrow[g] = lrow[g] * cr + rs;
            corr[g] = cr;
            int mr = g + mhi;
            Ps[w][mr][fr]      = f2bf(p0);
            Ps[w][mr][fr + 16] = f2bf(p1);
        }
        #pragma unroll
        for (int d8 = 0; d8 < 8; ++d8) {
            #pragma unroll
            for (int g = 0; g < 8; ++g) o[d8][g] *= corr[g];
        }
        v16bf aP = ld_frag(&Ps[w][fr][kh]);  // same-wave DS order guarantees RAW
        #pragma unroll
        for (int d8 = 0; d8 < 8; ++d8) {
            v16bf bV = ld_frag(V + (size_t)(d8 * 16 + fr) * T_SEQ + kb0 + kh);
            o[d8] = __builtin_amdgcn_wmma_f32_16x16x32_bf16(false, aP, false, bV,
                                                            (short)0, o[d8], false, false);
        }
    }

    #pragma unroll
    for (int g = 0; g < 8; ++g) {
        float inv = 1.0f / lrow[g];
        int tq = qw + g + mhi;
        #pragma unroll
        for (int d8 = 0; d8 < 8; ++d8)
            ob[(size_t)tq * C_DIM + h * HD + d8 * 16 + fr] = f2bf(o[d8][g] * inv);
    }
}

// ---------------------------------------------------------------------------
// Row-wise L2 normalization: dst[t] = src[t] / max(||src[t]||, 1e-12)
// ---------------------------------------------------------------------------
__global__ __launch_bounds__(256)
void l2norm_rows(const float* __restrict__ src, float* __restrict__ dst) {
    __shared__ float red[256];
    int t = blockIdx.x;
    const float* r = src + (size_t)t * C_DIM;
    float s = 0.f;
    for (int c = threadIdx.x; c < C_DIM; c += 256) { float v = r[c]; s += v * v; }
    red[threadIdx.x] = s;
    __syncthreads();
    for (int st = 128; st > 0; st >>= 1) {
        if (threadIdx.x < st) red[threadIdx.x] += red[threadIdx.x + st];
        __syncthreads();
    }
    float inv = 1.0f / fmaxf(sqrtf(red[0]), 1e-12f);
    for (int c = threadIdx.x; c < C_DIM; c += 256)
        dst[(size_t)t * C_DIM + c] = r[c] * inv;
}

// ---------------------------------------------------------------------------
extern "C" void kernel_launch(void* const* d_in, const int* in_sizes, int n_in,
                              void* d_out, int out_size, void* d_ws, size_t ws_size,
                              hipStream_t stream) {
    (void)in_sizes; (void)n_in; (void)out_size; (void)ws_size;
    const float* x   = (const float*)d_in[0];
    const float* wq  = (const float*)d_in[1];
    const float* wk  = (const float*)d_in[2];
    const float* wv  = (const float*)d_in[3];
    const float* wo  = (const float*)d_in[4];
    const float* sqk = (const float*)d_in[5];

    float* out   = (float*)d_out;                         // l2norm(out): [T][C]
    float* new_k = out   + (size_t)T_SEQ * C_DIM;         // [H][T][D]
    float* new_v = new_k + (size_t)T_SEQ * C_DIM;         // [H][T][D]

    char* ws = (char*)d_ws;
    const size_t NB16 = (size_t)T_SEQ * C_DIM * 2;        // 8 MiB per bf16 plane
    unsigned short* xb  = (unsigned short*)(ws);
    unsigned short* wqb = (unsigned short*)(ws + 1 * NB16);
    unsigned short* wkb = (unsigned short*)(ws + 2 * NB16);
    unsigned short* wvb = (unsigned short*)(ws + 3 * NB16);
    unsigned short* wob = (unsigned short*)(ws + 4 * NB16);
    unsigned short* qb2 = (unsigned short*)(ws + 5 * NB16);   // bf16 [h][t][d]
    unsigned short* kb2 = (unsigned short*)(ws + 6 * NB16);   // bf16 [h][t][d]
    unsigned short* vbt = (unsigned short*)(ws + 7 * NB16);   // bf16 [h][d][t]
    unsigned short* ab  = (unsigned short*)(ws + 8 * NB16);   // bf16 [t][C]
    float* q_pre = (float*)(ws + 9 * NB16);                   // f32 [h][t][d]
    float* o_pre = (float*)(ws + 9 * NB16 + (size_t)T_SEQ * C_DIM * 4);

    const int n4 = T_SEQ * C_DIM / 4;
    const int cg = (n4 + 255) / 256;
    cvt_f32_bf16<<<cg, 256, 0, stream>>>(x,  xb,  n4);
    cvt_f32_bf16<<<cg, 256, 0, stream>>>(wq, wqb, n4);
    cvt_f32_bf16<<<cg, 256, 0, stream>>>(wk, wkb, n4);
    cvt_f32_bf16<<<cg, 256, 0, stream>>>(wv, wvb, n4);
    cvt_f32_bf16<<<cg, 256, 0, stream>>>(wo, wob, n4);

    dim3 gg(T_SEQ / 128, C_DIM / 128);
    gemm_bf16_nt<<<gg, 256, 0, stream>>>(xb, wqb, q_pre, 2, nullptr, 0);
    gemm_bf16_nt<<<gg, 256, 0, stream>>>(xb, wkb, new_k, 2, nullptr, 0);  // pre-RoPE K out
    gemm_bf16_nt<<<gg, 256, 0, stream>>>(xb, wvb, new_v, 2, vbt, 3);      // V out + bf16 [h][d][t]

    rope_scale<<<(NH * T_SEQ * 64) / 256, 256, 0, stream>>>(q_pre, new_k, sqk, qb2, kb2);

    dim3 ga(T_SEQ / 128, NH);
    flash_attn<<<ga, 256, 0, stream>>>(qb2, kb2, vbt, ab);

    gemm_bf16_nt<<<gg, 256, 0, stream>>>(ab, wob, o_pre, 1, nullptr, 0);

    l2norm_rows<<<T_SEQ, 256, 0, stream>>>(o_pre, out);
}
